// Attention_68006512164912
// MI455X (gfx1250) — matrix operand order repaired and checked by reading, not compiled, not run on hardware
//
#include <hip/hip_runtime.h>
#include <hip/hip_bf16.h>

typedef _Float16 v16h __attribute__((ext_vector_type(16)));
typedef _Float16 h8   __attribute__((ext_vector_type(8)));
typedef float    v8f  __attribute__((ext_vector_type(8)));

#define DIMX     1024
#define HEADS    16
#define HEAD_DIM 64
#define NTOK     341
#define NPAD     384
#define BATCH    32
#define MROWS    (BATCH * NTOK)   // 10912
#define MTILES   (MROWS / 16)     // 682 (exact)
#define BHCNT    (BATCH * HEADS)  // 512
#define CHUNK    96               // kv tokens staged per LDS buffer
#define NCHUNK   4                // 4 * 96 = 384 = NPAD
#define NPAIR    3                // 32-token pairs per chunk

#if defined(__has_builtin)
#if __has_builtin(__builtin_amdgcn_global_load_async_to_lds_b128)
#define HAVE_ASYNC_LDS 1
#endif
#endif

// async-LDS builtin takes int4* in global (AS1) / LDS (AS3) address spaces
typedef int i4 __attribute__((vector_size(4 * sizeof(int))));
typedef __attribute__((address_space(1))) i4* gvec_t;
typedef __attribute__((address_space(3))) i4* lvec_t;

static __device__ __forceinline__ void async_copy16(const _Float16* g, _Float16* l) {
#if defined(HAVE_ASYNC_LDS)
  // 16B global -> LDS, tracked by ASYNCcnt (gfx1250 async-LDS path)
  __builtin_amdgcn_global_load_async_to_lds_b128((gvec_t)(void*)g, (lvec_t)(void*)l, 0, 0);
#else
  *(uint4*)l = *(const uint4*)g;
#endif
}

static __device__ __forceinline__ void wait_async_all() {
#if defined(HAVE_ASYNC_LDS)
#if __has_builtin(__builtin_amdgcn_s_wait_asynccnt)
  __builtin_amdgcn_s_wait_asynccnt(0);
#else
  asm volatile("s_wait_asynccnt 0" ::: "memory");
#endif
#endif
}

static __device__ __forceinline__ v16h make16(h8 lo, h8 hi) {
  return __builtin_shufflevector(lo, hi, 0, 1, 2, 3, 4, 5, 6, 7, 8, 9, 10, 11, 12, 13, 14, 15);
}
static __device__ __forceinline__ v8f wmma_f16(v16h a, v16h b, v8f c) {
  // D = A(16x32 f16) * B(32x16 f16) + C(16x16 f32)
  return __builtin_amdgcn_wmma_f32_16x16x32_f16(false, a, false, b, (short)0, c, false, false);
}

// Block-causal: last visible column (exclusive) for a given row.
static __device__ __forceinline__ int seg_end(int row) {
  if (row < 1) return 1;
  if (row < 5) return 5;
  if (row < 21) return 21;
  if (row < 85) return 85;
  return 341;
}

// ---------------------------------------------------------------------------
// 1) RoPE tables (reference builds these as module constants, not inputs)
// ---------------------------------------------------------------------------
__global__ void rope_kernel(float* __restrict__ cosT, float* __restrict__ sinT) {
  int idx = blockIdx.x * blockDim.x + threadIdx.x;
  if (idx >= NTOK * HEAD_DIM) return;
  int n = idx >> 6;
  int d = idx & 63;
  const int scales[5] = {1, 2, 4, 8, 16};
  int local = n;
  int s = 16;
#pragma unroll
  for (int i = 0; i < 5; ++i) {
    int sc = scales[i];
    int sz = sc * sc;
    if (local < sz) { s = sc; break; }
    local -= sz;
  }
  int y = local / s;
  int x = local - y * s;
  int t = (d < 32) ? y : x;  // first 32 dims: fy, next 32: fx
  int j = (d & 31) >> 1;     // frequency pair index 0..15
  float inv = __powf(10000.0f, -(float)j / 16.0f);
  float pos = (float)t * 16.0f / (float)s;  // scaled to pt grid
  float f = pos * inv;
  cosT[idx] = cosf(f);
  sinT[idx] = sinf(f);
}

// ---------------------------------------------------------------------------
// 2) f32 -> f16 conversion
// ---------------------------------------------------------------------------
__global__ void f32_to_f16_kernel(const float* __restrict__ src, _Float16* __restrict__ dst, int count) {
  int i = blockIdx.x * blockDim.x + threadIdx.x;
  if (i < count) dst[i] = (_Float16)src[i];
}

// ---------------------------------------------------------------------------
// 3) QKV GEMM (10912x1024 @ 1024x3072) + bias + RoPE, scatter to q/k/v layouts
//    Each wave: 16 rows x 64 cols, K loop of 32 -> 128 WMMA per wave.
// ---------------------------------------------------------------------------
__global__ __launch_bounds__(256) void qkv_gemm_kernel(
    const _Float16* __restrict__ Xh, const _Float16* __restrict__ Wh,
    const float* __restrict__ bias, const float* __restrict__ ropeC,
    const float* __restrict__ ropeS, _Float16* __restrict__ Qh,
    _Float16* __restrict__ Kh, _Float16* __restrict__ Vt) {
  int lane = threadIdx.x & 31;
  int wave = threadIdx.x >> 5;
  int rowTile = blockIdx.x * 8 + wave;
  if (rowTile >= MTILES) return;  // wave-uniform
  int m0 = rowTile * 16;
  int n0 = blockIdx.y * 64;
  int hi = lane >> 4;      // half-wave select
  int lo16 = lane & 15;

  v8f acc[4];
#pragma unroll
  for (int j = 0; j < 4; ++j)
#pragma unroll
    for (int r = 0; r < 8; ++r) acc[j][r] = 0.0f;

  // A fragment (contiguous k order, same as B): lanes 0-15 take K{k0..k0+15},
  // lanes 16-31 take K{k0+16..k0+31}; row = lane&15.
  const _Float16* arow = Xh + (size_t)(m0 + lo16) * DIMX + hi * 16;
  for (int k0 = 0; k0 < DIMX; k0 += 32) {
    __builtin_prefetch(arow + k0 + 256, 0, 1);
    h8 alo = *(const h8*)(arow + k0);
    h8 ahi = *(const h8*)(arow + k0 + 8);
    v16h a = make16(alo, ahi);
#pragma unroll
    for (int j = 0; j < 4; ++j) {
      // B fragment (32x16): col = lane&15, K range = k0 + (lane>>4)*16 .. +15.
      // B column m == row m of W (row-major [m][k]) -> contiguous 32B.
      const _Float16* bp = Wh + (size_t)(n0 + j * 16 + lo16) * DIMX + k0 + hi * 16;
      v16h b = *(const v16h*)bp;
      acc[j] = wmma_f16(a, b, acc[j]);
    }
  }

  // Epilogue: C layout -> element (r, lane) = [row m0+r+hi*8][col n0+j*16+lo16]
#pragma unroll
  for (int j = 0; j < 4; ++j) {
    int col = n0 + j * 16 + lo16;
    float bv = bias[col];
    int which = col >> 10;  // 0=q 1=k 2=v
    int crem = col & 1023;
    int h = crem >> 6;
    int d = crem & 63;
#pragma unroll
    for (int r = 0; r < 8; ++r) {
      int row = m0 + r + hi * 8;
      int bb = row / NTOK;
      int n = row - bb * NTOK;
      float v = acc[j][r] + bv;
      if (which == 2) {
        // V stored transposed: [BH][64][NPAD] so PV B-frags are contiguous
        Vt[((size_t)(bb * HEADS + h) * HEAD_DIM + d) * NPAD + n] = (_Float16)v;
      } else {
        float c = ropeC[n * HEAD_DIM + d];
        float s = ropeS[n * HEAD_DIM + d];
        // interleaved-pair partner lives one lane over (col ^ 1), same row
        float partner = __shfl_xor(v, 1, 32);
        float rot = (d & 1) ? partner : -partner;
        float o = v * c + rot * s;
        _Float16* dst = (which == 0) ? Qh : Kh;
        dst[((size_t)(bb * HEADS + h) * NPAD + n) * HEAD_DIM + d] = (_Float16)o;
      }
    }
  }
}

// ---------------------------------------------------------------------------
// 4) Flash attention: one block per (b,h); 8 waves; async double-buffered
//    KV chunks of 96 tokens in LDS; block-causal pair skipping.
// ---------------------------------------------------------------------------
static __device__ __forceinline__ void stage_chunk(
    const _Float16* __restrict__ Kh, const _Float16* __restrict__ Vt,
    int bh, int c, _Float16* ks, _Float16* vs, int tid) {
  // K chunk: [CHUNK][64] f16 = 768 x 16B packets; token = t>>3, off = (t&7)*8
  // V^T chunk: [64][CHUNK] f16 = 768 x 16B packets; d = t/12, off = (t%12)*8
  // Both land at linear LDS offset t*8 halves.
#pragma unroll
  for (int i = 0; i < 3; ++i) {
    int t = tid + i * 256;
    const _Float16* gk = Kh + ((size_t)bh * NPAD + c * CHUNK + (t >> 3)) * HEAD_DIM + (t & 7) * 8;
    async_copy16(gk, ks + t * 8);
    int d = t / 12;
    int o = t - d * 12;
    const _Float16* gv = Vt + ((size_t)bh * HEAD_DIM + d) * NPAD + c * CHUNK + o * 8;
    async_copy16(gv, vs + t * 8);
  }
}

__global__ __launch_bounds__(256) void attn_kernel(
    const _Float16* __restrict__ Qh, const _Float16* __restrict__ Kh,
    const _Float16* __restrict__ Vt, const float* __restrict__ mask,
    _Float16* __restrict__ Oh) {
  __shared__ _Float16 Ks[2][CHUNK * HEAD_DIM];  // 2 x 12KB [token][d]
  __shared__ _Float16 Vs[2][HEAD_DIM * CHUNK];  // 2 x 12KB [d][token]
  __shared__ _Float16 Ps[8][16 * 32];           // per-wave P scratch 8KB

  int bh = blockIdx.x;
  int b = bh >> 4;
  int h = bh & 15;
  int lane = threadIdx.x & 31;
  int wave = threadIdx.x >> 5;
  int hi = lane >> 4;
  int lo16 = lane & 15;

  v8f O[3][4];
  float mrow[3][8], lrow[3][8];
#pragma unroll
  for (int qi = 0; qi < 3; ++qi) {
#pragma unroll
    for (int j = 0; j < 4; ++j)
#pragma unroll
      for (int r = 0; r < 8; ++r) O[qi][j][r] = 0.0f;
#pragma unroll
    for (int r = 0; r < 8; ++r) { mrow[qi][r] = -1e30f; lrow[qi][r] = 0.0f; }
  }

  // prologue: fill buffer 0
  stage_chunk(Kh, Vt, bh, 0, Ks[0], Vs[0], threadIdx.x);
  wait_async_all();
  __syncthreads();

  for (int c = 0; c < NCHUNK; ++c) {
    int buf = c & 1;
    if (c + 1 < NCHUNK)  // overlap next chunk's fill with this chunk's WMMA
      stage_chunk(Kh, Vt, bh, c + 1, Ks[buf ^ 1], Vs[buf ^ 1], threadIdx.x);

#pragma unroll
    for (int qi = 0; qi < 3; ++qi) {
      int qt = wave + qi * 8;  // query tile id (16 rows each)
      if (qt >= 22) continue;  // wave-uniform; EXEC stays all-ones
      int E = seg_end(qt * 16 + 15 < NTOK ? qt * 16 + 15 : NTOK - 1);
      if (c * CHUNK >= E) continue;  // entire chunk block-causally masked

      const _Float16* qrow = Qh + ((size_t)bh * NPAD + qt * 16 + lo16) * HEAD_DIM + hi * 16;
      v16h a0 = make16(*(const h8*)(qrow), *(const h8*)(qrow + 8));       // d 0..31
      v16h a1 = make16(*(const h8*)(qrow + 32), *(const h8*)(qrow + 40));  // d 32..63

#pragma unroll
      for (int p = 0; p < NPAIR; ++p) {  // 32 kv tokens per step
        if (c * CHUNK + p * 32 >= E) continue;  // masked pair (wave-uniform)
        v8f s0, s1;
        {
          const _Float16* kp = Ks[buf] + (p * 32 + lo16) * HEAD_DIM + hi * 16;
          v16h b0 = *(const v16h*)kp;
          v16h b1 = *(const v16h*)(kp + 32);
          v8f z = {0.f, 0.f, 0.f, 0.f, 0.f, 0.f, 0.f, 0.f};
          z = wmma_f16(a0, b0, z);
          s0 = wmma_f16(a1, b1, z);
        }
        {
          const _Float16* kp = Ks[buf] + (p * 32 + 16 + lo16) * HEAD_DIM + hi * 16;
          v16h b0 = *(const v16h*)kp;
          v16h b1 = *(const v16h*)(kp + 32);
          v8f z = {0.f, 0.f, 0.f, 0.f, 0.f, 0.f, 0.f, 0.f};
          z = wmma_f16(a0, b0, z);
          s1 = wmma_f16(a1, b1, z);
        }
        // scale + block-causal mask (from input tensor)
        int col0 = c * CHUNK + p * 32 + lo16;
        int col1 = col0 + 16;
        int rowbase = qt * 16 + hi * 8;
#pragma unroll
        for (int r = 0; r < 8; ++r) {
          int row = rowbase + r;
          int rm = (row < NTOK) ? row : (NTOK - 1);
          float v0 = s0[r] * 0.125f;
          float v1 = s1[r] * 0.125f;
          v0 = (col0 < NTOK) ? (v0 + mask[(size_t)rm * NTOK + col0]) : -1e9f;
          v1 = (col1 < NTOK) ? (v1 + mask[(size_t)rm * NTOK + col1]) : -1e9f;
          s0[r] = v0;
          s1[r] = v1;
        }
        // flash update; row stats per C-layout slot, reduced across 16 lanes
        float alpha[8];
#pragma unroll
        for (int r = 0; r < 8; ++r) {
          float mt = fmaxf(s0[r], s1[r]);
          mt = fmaxf(mt, __shfl_xor(mt, 1, 32));
          mt = fmaxf(mt, __shfl_xor(mt, 2, 32));
          mt = fmaxf(mt, __shfl_xor(mt, 4, 32));
          mt = fmaxf(mt, __shfl_xor(mt, 8, 32));
          float mn = fmaxf(mrow[qi][r], mt);
          alpha[r] = __expf(mrow[qi][r] - mn);
          mrow[qi][r] = mn;
          float p0 = __expf(s0[r] - mn);
          float p1 = __expf(s1[r] - mn);
          float rs = p0 + p1;
          rs += __shfl_xor(rs, 1, 32);
          rs += __shfl_xor(rs, 2, 32);
          rs += __shfl_xor(rs, 4, 32);
          rs += __shfl_xor(rs, 8, 32);
          lrow[qi][r] = lrow[qi][r] * alpha[r] + rs;
          // C layout -> LDS so we can reload as an A fragment
          Ps[wave][(r + hi * 8) * 32 + lo16] = (_Float16)p0;
          Ps[wave][(r + hi * 8) * 32 + 16 + lo16] = (_Float16)p1;
        }
        // same-wave LDS is in-order; wait for stores, then reload in A layout
        asm volatile("s_wait_dscnt 0" ::: "memory");
        const _Float16* pp = &Ps[wave][lo16 * 32 + hi * 16];
        v16h pa = make16(*(const h8*)pp, *(const h8*)(pp + 8));
#pragma unroll
        for (int j = 0; j < 4; ++j) {
          // B frag of V: col d = j*16+lo16, k tokens = p*32 + hi*16 .. +15 (contiguous in V^T)
          const _Float16* vp = Vs[buf] + (j * 16 + lo16) * CHUNK + p * 32 + hi * 16;
          v16h vb = *(const v16h*)vp;
          v8f o = O[qi][j];
#pragma unroll
          for (int r = 0; r < 8; ++r) o[r] *= alpha[r];
          O[qi][j] = wmma_f16(pa, vb, o);
        }
      }
    }
    // next buffer must be fully landed and this chunk fully consumed block-wide
    wait_async_all();
    __syncthreads();
  }

  // normalize + store [B*N][DIM] f16 (input to proj GEMM)
#pragma unroll
  for (int qi = 0; qi < 3; ++qi) {
    int qt = wave + qi * 8;
    if (qt >= 22) continue;
#pragma unroll
    for (int r = 0; r < 8; ++r) {
      int row = qt * 16 + hi * 8 + r;
      if (row >= NTOK) continue;
      float inv = 1.0f / lrow[qi][r];
#pragma unroll
      for (int j = 0; j < 4; ++j) {
        int d = j * 16 + lo16;
        float o = O[qi][j][r] * inv;
        Oh[((size_t)(b * NTOK + row)) * DIMX + h * HEAD_DIM + d] = (_Float16)o;
      }
    }
  }
}

// ---------------------------------------------------------------------------
// 5) Output projection GEMM (10912x1024 @ 1024x1024) + bias, f32 out
// ---------------------------------------------------------------------------
__global__ __launch_bounds__(256) void proj_gemm_kernel(
    const _Float16* __restrict__ Ah, const _Float16* __restrict__ Wh,
    const float* __restrict__ bias, float* __restrict__ out) {
  int lane = threadIdx.x & 31;
  int wave = threadIdx.x >> 5;
  int rowTile = blockIdx.x * 8 + wave;
  if (rowTile >= MTILES) return;
  int m0 = rowTile * 16;
  int n0 = blockIdx.y * 64;
  int hi = lane >> 4;
  int lo16 = lane & 15;

  v8f acc[4];
#pragma unroll
  for (int j = 0; j < 4; ++j)
#pragma unroll
    for (int r = 0; r < 8; ++r) acc[j][r] = 0.0f;

  const _Float16* arow = Ah + (size_t)(m0 + lo16) * DIMX + hi * 16;
  for (int k0 = 0; k0 < DIMX; k0 += 32) {
    __builtin_prefetch(arow + k0 + 256, 0, 1);
    h8 alo = *(const h8*)(arow + k0);
    h8 ahi = *(const h8*)(arow + k0 + 8);
    v16h a = make16(alo, ahi);
#pragma unroll
    for (int j = 0; j < 4; ++j) {
      const _Float16* bp = Wh + (size_t)(n0 + j * 16 + lo16) * DIMX + k0 + hi * 16;
      v16h b = *(const v16h*)bp;
      acc[j] = wmma_f16(a, b, acc[j]);
    }
  }

#pragma unroll
  for (int j = 0; j < 4; ++j) {
    int col = n0 + j * 16 + lo16;
    float bv = bias[col];
#pragma unroll
    for (int r = 0; r < 8; ++r) {
      int row = m0 + r + hi * 8;
      out[(size_t)row * DIMX + col] = acc[j][r] + bv;
    }
  }
}

// ---------------------------------------------------------------------------
extern "C" void kernel_launch(void* const* d_in, const int* in_sizes, int n_in,
                              void* d_out, int out_size, void* d_ws, size_t ws_size,
                              hipStream_t stream) {
  const float* x = (const float*)d_in[0];
  const float* mask = (const float*)d_in[1];
  const float* qkv_w = (const float*)d_in[2];
  const float* qkv_b = (const float*)d_in[3];
  const float* proj_w = (const float*)d_in[4];
  const float* proj_b = (const float*)d_in[5];
  float* out = (float*)d_out;

  char* ws = (char*)d_ws;
  size_t off = 0;
  auto alloc = [&](size_t bytes) -> void* {
    void* p = ws + off;
    off = (off + bytes + 255) & ~(size_t)255;
    return p;
  };

  float* ropeC = (float*)alloc((size_t)NTOK * HEAD_DIM * 4);
  float* ropeS = (float*)alloc((size_t)NTOK * HEAD_DIM * 4);
  _Float16* Xh = (_Float16*)alloc((size_t)MROWS * DIMX * 2);
  _Float16* Wqkv = (_Float16*)alloc((size_t)3 * DIMX * DIMX * 2);
  _Float16* Wproj = (_Float16*)alloc((size_t)DIMX * DIMX * 2);
  _Float16* Qh = (_Float16*)alloc((size_t)BHCNT * NPAD * HEAD_DIM * 2);
  _Float16* Kh = (_Float16*)alloc((size_t)BHCNT * NPAD * HEAD_DIM * 2);
  _Float16* Vt = (_Float16*)alloc((size_t)BHCNT * HEAD_DIM * NPAD * 2);
  _Float16* Ah = (_Float16*)alloc((size_t)MROWS * DIMX * 2);

  rope_kernel<<<(NTOK * HEAD_DIM + 255) / 256, 256, 0, stream>>>(ropeC, ropeS);

  int cx = MROWS * DIMX;
  f32_to_f16_kernel<<<(cx + 255) / 256, 256, 0, stream>>>(x, Xh, cx);
  int cw = 3 * DIMX * DIMX;
  f32_to_f16_kernel<<<(cw + 255) / 256, 256, 0, stream>>>(qkv_w, Wqkv, cw);
  int cp = DIMX * DIMX;
  f32_to_f16_kernel<<<(cp + 255) / 256, 256, 0, stream>>>(proj_w, Wproj, cp);

  dim3 g1((MTILES + 7) / 8, 3 * DIMX / 64);  // (86, 48)
  qkv_gemm_kernel<<<g1, 256, 0, stream>>>(Xh, Wqkv, qkv_b, ropeC, ropeS, Qh, Kh, Vt);

  attn_kernel<<<BHCNT, 256, 0, stream>>>(Qh, Kh, Vt, mask, Ah);

  dim3 g2((MTILES + 7) / 8, DIMX / 64);  // (86, 16)
  proj_gemm_kernel<<<g2, 256, 0, stream>>>(Ah, Wproj, proj_b, out);
}
